// StageBranchRunnerN2_35046933135694
// MI455X (gfx1250) — matrix operand order repaired and checked by reading, not compiled, not run on hardware
//
#include <hip/hip_runtime.h>
#include <math.h>

typedef __attribute__((ext_vector_type(16))) _Float16 v16h;
typedef __attribute__((ext_vector_type(8)))  float    v8f;

#define D_MODEL   1024
#define RAW_FEAT  512
#define PROJ_DIM  256
#define R_IN      1280
#define R_HID     256
#define E_HID     256
#define NTOK      8192
#define NTILE     (NTOK / 16)

// ---------------------------------------------------------------- helpers ----

__device__ __forceinline__ float gelu_f(float x) {
  const float k = 0.7978845608028654f;  // sqrt(2/pi)
  return 0.5f * x * (1.0f + tanhf(k * (x + 0.044715f * x * x * x)));
}

// A fragment: 16x32 f16 tile, row-major in LDS, leading dim == K of the GEMM.
// Layout: lanes 0-15 row M=lane hold K[0..15]; lanes 16-31 hold K[16..31]
// (matches B fragment order) -> two adjacent 16-byte chunks.
__device__ __forceinline__ v16h load_a_f16(const _Float16* A, int lda, int k0) {
  const int lane = threadIdx.x & 31;
  const int row  = lane & 15;
  const int hi   = lane >> 4;
  const _Float16* p = A + row * lda + k0 + hi * 16;
  v16h a;
  ((float4*)&a)[0] = *(const float4*)(p);
  ((float4*)&a)[1] = *(const float4*)(p + 8);
  return a;
}

// B fragment: 32x16 f16 from pre-transposed weight BT[N][K].
// Layout (contiguous k, same as A): lane n (0-15) holds K[0..15] of column n, lanes 16-31 K[16..31]
// -> one contiguous 32-byte chunk per lane.
__device__ __forceinline__ v16h load_b_bt(const _Float16* BT, int K, int nbase, int k0) {
  const int lane = threadIdx.x & 31;
  const int n    = lane & 15;
  const int hi   = lane >> 4;
  return *(const v16h*)(BT + (size_t)(nbase + n) * K + k0 + hi * 16);
}

// One 16x16 output tile of A(16xK, LDS, lda==K) x BT(NxK, global), f32 acc.
// Software-pipelined one stage; unconditional speculative prefetch (safe: a
// failing/oob prefetch is silently dropped and our buffers sit inside a large
// mapped workspace).
template <int K>
__device__ __forceinline__ v8f wmma_rowtile(const _Float16* A, const _Float16* BT, int nbase) {
  const int n = threadIdx.x & 15;
  const _Float16* brow = BT + (size_t)(nbase + n) * K;
  v8f acc = {};
  v16h a = load_a_f16(A, K, 0);
  v16h b = load_b_bt(BT, K, nbase, 0);
  #pragma unroll 4
  for (int k0 = 32; k0 < K; k0 += 32) {
    __builtin_prefetch(brow + k0 + 224, 0, 1);
    v16h a1 = load_a_f16(A, K, k0);
    v16h b1 = load_b_bt(BT, K, nbase, k0);
    acc = __builtin_amdgcn_wmma_f32_16x16x32_f16(false, a, false, b, (short)0, acc,
                                                 false, false);
    a = a1;
    b = b1;
  }
  return __builtin_amdgcn_wmma_f32_16x16x32_f16(false, a, false, b, (short)0, acc,
                                                false, false);
}

__device__ __forceinline__ float block_sum256(float v, float* red) {
  #pragma unroll
  for (int off = 16; off > 0; off >>= 1) v += __shfl_xor(v, off, 32);
  if ((threadIdx.x & 31) == 0) red[threadIdx.x >> 5] = v;
  __syncthreads();
  float s = 0.f;
  #pragma unroll
  for (int i = 0; i < 8; ++i) s += red[i];
  __syncthreads();
  return s;
}

// ---------------------------------------------------------------- kernels ----

// src[K][N] f32 -> dst[Npad][K] f16 (transposed, zero-padded cols); batched in y.
__global__ void k_transpose_w(const float* __restrict__ src, _Float16* __restrict__ dst,
                              int K, int N, int Npad) {
  src += (size_t)blockIdx.y * K * N;
  dst += (size_t)blockIdx.y * Npad * K;
  int idx = blockIdx.x * blockDim.x + threadIdx.x;
  if (idx >= Npad * K) return;
  int n = idx / K, k = idx - n * K;
  float v = (n < N) ? src[(size_t)k * N + n] : 0.0f;
  dst[idx] = (_Float16)v;
}

__global__ __launch_bounds__(256) void k_layernorm(const float* __restrict__ hidden,
                                                   const float* __restrict__ g,
                                                   const float* __restrict__ b,
                                                   _Float16* __restrict__ hn) {
  __shared__ float red[8];
  const int t = blockIdx.x;
  const float* x = hidden + (size_t)t * D_MODEL;
  float v[4];
  float s = 0.f;
  #pragma unroll
  for (int i = 0; i < 4; ++i) { v[i] = x[threadIdx.x + i * 256]; s += v[i]; }
  s = block_sum256(s, red);
  const float mu = s * (1.0f / D_MODEL);
  float q = 0.f;
  #pragma unroll
  for (int i = 0; i < 4; ++i) { float d = v[i] - mu; q += d * d; }
  q = block_sum256(q, red);
  const float rs = rsqrtf(q * (1.0f / D_MODEL) + 1e-5f);
  #pragma unroll
  for (int i = 0; i < 4; ++i) {
    int c = threadIdx.x + i * 256;
    hn[(size_t)t * D_MODEL + c] = (_Float16)((v[i] - mu) * rs * g[c] + b[c]);
  }
}

// proj = gelu(bank x W1 + b1) x W2 + b2   (per 16-token tile)
__global__ __launch_bounds__(256) void k_featproj(const float* __restrict__ bank,
                                                  const _Float16* __restrict__ w1T,
                                                  const float* __restrict__ b1,
                                                  const _Float16* __restrict__ w2T,
                                                  const float* __restrict__ b2,
                                                  _Float16* __restrict__ proj) {
  __shared__ alignas(16) _Float16 Af[16 * RAW_FEAT];
  __shared__ alignas(16) _Float16 Hf[16 * PROJ_DIM];
  const int m0 = blockIdx.x * 16;
  // tile is 16*512 contiguous floats; convert to f16 in LDS
  const float4* src = (const float4*)(bank + (size_t)m0 * RAW_FEAT);
  for (int c = threadIdx.x; c < 16 * RAW_FEAT / 4; c += 256) {
    float4 f = src[c];
    _Float16* d = Af + c * 4;
    d[0] = (_Float16)f.x; d[1] = (_Float16)f.y; d[2] = (_Float16)f.z; d[3] = (_Float16)f.w;
  }
  __syncthreads();
  const int wave = threadIdx.x >> 5, lane = threadIdx.x & 31;
  const int n = lane & 15, hi = lane >> 4;
  #pragma unroll
  for (int t = 0; t < 2; ++t) {
    const int nbase = (wave * 2 + t) * 16;
    v8f acc = wmma_rowtile<RAW_FEAT>(Af, w1T, nbase);
    const float bias = b1[nbase + n];
    #pragma unroll
    for (int vv = 0; vv < 8; ++vv)
      Hf[(vv + 8 * hi) * PROJ_DIM + nbase + n] = (_Float16)gelu_f(acc[vv] + bias);
  }
  __syncthreads();
  #pragma unroll
  for (int t = 0; t < 2; ++t) {
    const int nbase = (wave * 2 + t) * 16;
    v8f acc = wmma_rowtile<PROJ_DIM>(Hf, w2T, nbase);
    const float bias = b2[nbase + n];
    #pragma unroll
    for (int vv = 0; vv < 8; ++vv)
      proj[(size_t)(m0 + vv + 8 * hi) * PROJ_DIM + nbase + n] = (_Float16)(acc[vv] + bias);
  }
}

// router: logits = gelu([hn|proj] x rW1 + b1) x rW2 + b2 ; top-2 softmax; rule logits
__global__ __launch_bounds__(256) void k_router(const _Float16* __restrict__ hn,
                                                const _Float16* __restrict__ proj,
                                                const _Float16* __restrict__ w1T,
                                                const float* __restrict__ b1,
                                                const _Float16* __restrict__ w2T,
                                                const float* __restrict__ b2,
                                                const float* __restrict__ feat,
                                                const float* __restrict__ rr_w,
                                                const float* __restrict__ rr_b,
                                                float* __restrict__ out_gate,
                                                float* __restrict__ out_scaled,
                                                float* __restrict__ out_group,
                                                float* __restrict__ out_rule) {
  __shared__ alignas(16) _Float16 Rin[16 * R_IN];
  __shared__ alignas(16) _Float16 Hr[16 * R_HID];
  __shared__ float Lg[16 * 8];
  const int m0 = blockIdx.x * 16;
  for (int c = threadIdx.x; c < 16 * R_IN / 8; c += 256) {   // 8-half (16B) chunks
    const int row = c / (R_IN / 8);
    const int off = (c - row * (R_IN / 8)) * 8;
    const _Float16* s = (off < D_MODEL)
        ? hn + (size_t)(m0 + row) * D_MODEL + off
        : proj + (size_t)(m0 + row) * PROJ_DIM + (off - D_MODEL);
    *(float4*)(Rin + row * R_IN + off) = *(const float4*)s;
  }
  __syncthreads();
  const int wave = threadIdx.x >> 5, lane = threadIdx.x & 31;
  const int n = lane & 15, hi = lane >> 4;
  #pragma unroll
  for (int t = 0; t < 2; ++t) {
    const int nbase = (wave * 2 + t) * 16;
    v8f acc = wmma_rowtile<R_IN>(Rin, w1T, nbase);
    const float bias = b1[nbase + n];
    #pragma unroll
    for (int vv = 0; vv < 8; ++vv)
      Hr[(vv + 8 * hi) * R_HID + nbase + n] = (_Float16)gelu_f(acc[vv] + bias);
  }
  __syncthreads();
  if (wave == 0) {                       // N padded to 16 (cols 8..15 are zero weights)
    v8f acc = wmma_rowtile<R_HID>(Hr, w2T, 0);
    if (n < 8) {
      const float bias = b2[n];
      #pragma unroll
      for (int vv = 0; vv < 8; ++vv) {
        const int m = vv + 8 * hi;
        const float lg = acc[vv] + bias;          // TEMP == 1
        Lg[m * 8 + n] = lg;
        out_scaled[(size_t)(m0 + m) * 8 + n] = lg;
      }
    }
  }
  __syncthreads();
  if (threadIdx.x < 16) {
    const int m = threadIdx.x, tk = m0 + m;
    float l[8], m1 = -3.4e38f, m2 = -3.4e38f;
    #pragma unroll
    for (int j = 0; j < 8; ++j) {
      l[j] = Lg[m * 8 + j];
      if (l[j] > m1) { m2 = m1; m1 = l[j]; } else if (l[j] > m2) m2 = l[j];
    }
    float w[8], s = 0.f;
    #pragma unroll
    for (int j = 0; j < 8; ++j) { w[j] = (l[j] >= m2) ? expf(l[j] - m1) : 0.f; s += w[j]; }
    const float inv = 1.0f / s;
    #pragma unroll
    for (int j = 0; j < 8; ++j) out_gate[(size_t)tk * 8 + j] = w[j] * inv;
    #pragma unroll
    for (int gidx = 0; gidx < 4; ++gidx)
      out_group[(size_t)tk * 4 + gidx] = (w[2 * gidx] + w[2 * gidx + 1]) * inv;
    #pragma unroll
    for (int j = 0; j < 8; ++j) {
      float r = rr_b[j];
      for (int i = 0; i < 16; ++i) r += feat[(size_t)tk * 16 + i] * rr_w[i * 8 + j];
      out_rule[(size_t)tk * 8 + j] = r;           // TEMP == 1
    }
  }
}

// experts: stage_out = sum_e gate_e*(gelu(hn x W1_e + b1_e) x W2_e + b2_e);
// next_hidden = hidden + alpha*stage_out. Zero-gate experts skipped per tile.
__global__ __launch_bounds__(256) void k_experts(const float* __restrict__ hidden,
                                                 const _Float16* __restrict__ hn,
                                                 const _Float16* __restrict__ w1T,
                                                 const float* __restrict__ b1,
                                                 const _Float16* __restrict__ w2T,
                                                 const float* __restrict__ b2,
                                                 const float* __restrict__ gate,
                                                 const float* __restrict__ alpha_p,
                                                 float* __restrict__ out_next,
                                                 float* __restrict__ out_stage) {
  __shared__ alignas(16) _Float16 Hn[16 * D_MODEL];   // 32 KB
  __shared__ alignas(16) _Float16 H1[16 * E_HID];     //  8 KB
  __shared__ float Gt[16 * 8];
  __shared__ float Gsum[8];
  const int m0 = blockIdx.x * 16;
  const float4* src = (const float4*)(hn + (size_t)m0 * D_MODEL);
  float4* dst = (float4*)Hn;
  for (int c = threadIdx.x; c < 16 * D_MODEL / 8; c += 256) dst[c] = src[c];
  if (threadIdx.x < 128) Gt[threadIdx.x] = gate[(size_t)m0 * 8 + threadIdx.x];
  __syncthreads();
  if (threadIdx.x < 8) {
    float s = 0.f;
    for (int m = 0; m < 16; ++m) s += Gt[m * 8 + threadIdx.x];
    Gsum[threadIdx.x] = s;
  }
  __syncthreads();
  const int wave = threadIdx.x >> 5, lane = threadIdx.x & 31;
  const int n = lane & 15, hi = lane >> 4;
  v8f accO[8];
  #pragma unroll
  for (int j = 0; j < 8; ++j)
    #pragma unroll
    for (int vv = 0; vv < 8; ++vv) accO[j][vv] = 0.f;

  for (int e = 0; e < 8; ++e) {
    if (Gsum[e] == 0.f) continue;        // block-uniform; barriers skipped together
    const _Float16* W1 = w1T + (size_t)e * E_HID * D_MODEL;   // [256][1024]
    const _Float16* W2 = w2T + (size_t)e * D_MODEL * E_HID;   // [1024][256]
    #pragma unroll
    for (int t = 0; t < 2; ++t) {
      const int nbase = (wave * 2 + t) * 16;
      v8f acc = wmma_rowtile<D_MODEL>(Hn, W1, nbase);
      const float bias = b1[e * E_HID + nbase + n];
      #pragma unroll
      for (int vv = 0; vv < 8; ++vv)
        H1[(vv + 8 * hi) * E_HID + nbase + n] = (_Float16)gelu_f(acc[vv] + bias);
    }
    __syncthreads();
    // GEMM2: hoist the 8 A-fragments of the shared 16x256 H1 tile into
    // registers once, then stream only B across the wave's 8 N-tiles
    // (64 back-to-back WMMAs, no redundant LDS traffic).
    v16h afr[8];
    #pragma unroll
    for (int kk = 0; kk < 8; ++kk) afr[kk] = load_a_f16(H1, E_HID, kk * 32);
    #pragma unroll
    for (int j = 0; j < 8; ++j) {
      const int nbase = (wave * 8 + j) * 16;
      v8f acc = {};
      #pragma unroll
      for (int kk = 0; kk < 8; ++kk) {
        v16h b = load_b_bt(W2, E_HID, nbase, kk * 32);
        acc = __builtin_amdgcn_wmma_f32_16x16x32_f16(false, afr[kk], false, b, (short)0,
                                                     acc, false, false);
      }
      const float bias = b2[e * D_MODEL + nbase + n];
      #pragma unroll
      for (int vv = 0; vv < 8; ++vv) {
        const float g = Gt[(vv + 8 * hi) * 8 + e];
        accO[j][vv] += g * (acc[vv] + bias);
      }
    }
    __syncthreads();
  }
  const float alpha = alpha_p[0];
  #pragma unroll
  for (int j = 0; j < 8; ++j) {
    const int nbase = (wave * 8 + j) * 16;
    #pragma unroll
    for (int vv = 0; vv < 8; ++vv) {
      const int m = vv + 8 * hi;
      const size_t idx = (size_t)(m0 + m) * D_MODEL + nbase + n;
      const float so = accO[j][vv];
      out_stage[idx] = so;
      out_next[idx] = hidden[idx] + alpha * so;
    }
  }
}

// ---------------------------------------------------------------- launch ----

extern "C" void kernel_launch(void* const* d_in, const int* in_sizes, int n_in,
                              void* d_out, int out_size, void* d_ws, size_t ws_size,
                              hipStream_t stream) {
  const float* hidden = (const float*)d_in[0];
  const float* feat   = (const float*)d_in[1];
  const float* bank   = (const float*)d_in[2];
  // d_in[3] item_seq_len: unused by the reference math
  const float* ln_g   = (const float*)d_in[4];
  const float* ln_b   = (const float*)d_in[5];
  const float* fp_w1  = (const float*)d_in[6];
  const float* fp_b1  = (const float*)d_in[7];
  const float* fp_w2  = (const float*)d_in[8];
  const float* fp_b2  = (const float*)d_in[9];
  const float* r_w1   = (const float*)d_in[10];
  const float* r_b1   = (const float*)d_in[11];
  const float* r_w2   = (const float*)d_in[12];
  const float* r_b2   = (const float*)d_in[13];
  const float* rr_w   = (const float*)d_in[14];
  const float* rr_b   = (const float*)d_in[15];
  const float* e_w1   = (const float*)d_in[16];
  const float* e_b1   = (const float*)d_in[17];
  const float* e_w2   = (const float*)d_in[18];
  const float* e_b2   = (const float*)d_in[19];
  const float* alpha  = (const float*)d_in[20];

  char* ws = (char*)d_ws;
  _Float16* hn    = (_Float16*)(ws + 0);          // 8192*1024 f16   = 16 MB
  _Float16* projw = (_Float16*)(ws + 16777216);   // 8192*256  f16   =  4 MB
  _Float16* fpw1T = (_Float16*)(ws + 20971520);   // [256][512]
  _Float16* fpw2T = (_Float16*)(ws + 21233664);   // [256][256]
  _Float16* rw1T  = (_Float16*)(ws + 21364736);   // [256][1280]
  _Float16* rw2T  = (_Float16*)(ws + 22020096);   // [16][256] (N padded 8->16)
  _Float16* ew1T  = (_Float16*)(ws + 22028288);   // [8][256][1024]
  _Float16* ew2T  = (_Float16*)(ws + 26222592);   // [8][1024][256]

  float* out        = (float*)d_out;
  float* out_next   = out;             // (4,2048,1024)
  float* out_stage  = out + 8388608;   // (4,2048,1024)
  float* out_gate   = out + 16777216;  // (4,2048,8)
  float* out_scaled = out + 16842752;  // (4,2048,8)
  float* out_group  = out + 16908288;  // (4,2048,4)
  float* out_rule   = out + 16941056;  // (4,2048,8)

  const dim3 blk(256);
  k_transpose_w<<<dim3((256 * 512 + 255) / 256, 1), blk, 0, stream>>>(fp_w1, fpw1T, 512, 256, 256);
  k_transpose_w<<<dim3((256 * 256 + 255) / 256, 1), blk, 0, stream>>>(fp_w2, fpw2T, 256, 256, 256);
  k_transpose_w<<<dim3((256 * 1280 + 255) / 256, 1), blk, 0, stream>>>(r_w1, rw1T, 1280, 256, 256);
  k_transpose_w<<<dim3((16 * 256 + 255) / 256, 1), blk, 0, stream>>>(r_w2, rw2T, 256, 8, 16);
  k_transpose_w<<<dim3((256 * 1024 + 255) / 256, 8), blk, 0, stream>>>(e_w1, ew1T, 1024, 256, 256);
  k_transpose_w<<<dim3((1024 * 256 + 255) / 256, 8), blk, 0, stream>>>(e_w2, ew2T, 256, 1024, 1024);

  k_layernorm<<<NTOK, blk, 0, stream>>>(hidden, ln_g, ln_b, hn);
  k_featproj<<<NTILE, blk, 0, stream>>>(bank, fpw1T, fp_b1, fpw2T, fp_b2, projw);
  k_router<<<NTILE, blk, 0, stream>>>(hn, projw, rw1T, r_b1, rw2T, r_b2, feat, rr_w, rr_b,
                                      out_gate, out_scaled, out_group, out_rule);
  k_experts<<<NTILE, blk, 0, stream>>>(hidden, hn, ew1T, e_b1, ew2T, e_b2,
                                       out_gate, alpha, out_next, out_stage);

  (void)in_sizes; (void)n_in; (void)out_size; (void)ws_size;
}
